// DistLayer_88794153877519
// MI455X (gfx1250) — matrix operand order repaired and checked by reading, not compiled, not run on hardware
//
#include <hip/hip_runtime.h>
#include <hip/hip_bf16.h>

typedef __attribute__((ext_vector_type(16))) __bf16 v16bf;
typedef __attribute__((ext_vector_type(8)))  __bf16 v8bf;
typedef __attribute__((ext_vector_type(8)))  float  v8f;

#define NROWS 200000
#define NAE   128
#define CDIM  256        // 2*NAE
#define KDIM  384        // NDE + 2*NAE
#define GSEG  1024
#define ESEG  100
#define MTILE 64
#define ASTRIDE 392      // padded bf16 row stride (16B aligned, conflict-free)

// ---------------- ws layout (bytes) ----------------
#define OFF_ATOM_SUMS 0u          // 1024*128 f32 = 524288
#define OFF_ATOM_CNT  524288u     // 1024 f32
#define OFF_ELE_SUMS  528384u     // 100*128 f32 = 51200
#define OFF_ELE_CNT   579584u     // 100 f32 (pad 512)
#define OFF_COLS      580096u     // 256 f32
#define OFF_COLSS     581120u     // 256 f32
#define ZERO_FLOATS   145536      // [0, 582144) zeroed each call
#define OFF_MU        582144u
#define OFF_RSTD      583168u
#define OFF_PA        584192u     // 1024*128 bf16 = 262144
#define OFF_PE        846336u     // 100*128 bf16 (pad 32768)
#define OFF_WP        879104u     // 384*256 bf16 = 196608 (32B aligned)

__global__ void zero_ws_kernel(float* p, int n) {
    int i = blockIdx.x * blockDim.x + threadIdx.x;
    int stride = gridDim.x * blockDim.x;
    for (; i < n; i += stride) p[i] = 0.0f;
}

// one block handles 2 rows: tid>>7 selects row, tid&127 selects column
__global__ void pool_accum_kernel(const float* __restrict__ x,
                                  const int* __restrict__ aidx,
                                  const int* __restrict__ eidx,
                                  float* __restrict__ asum, float* __restrict__ acnt,
                                  float* __restrict__ esum, float* __restrict__ ecnt) {
    int r = blockIdx.x * 2 + (threadIdx.x >> 7);
    int c = threadIdx.x & 127;
    if (r >= NROWS) return;
    int a = aidx[r], e = eidx[r];
    atomicAdd(&asum[a * NAE + c], x[(long)r * CDIM + c]);
    atomicAdd(&esum[e * NAE + c], x[(long)r * CDIM + NAE + c]);
    if (c == 0) { atomicAdd(&acnt[a], 1.0f); atomicAdd(&ecnt[e], 1.0f); }
}

__global__ void pool_final_kernel(const float* __restrict__ asum, const float* __restrict__ acnt,
                                  const float* __restrict__ esum, const float* __restrict__ ecnt,
                                  __bf16* __restrict__ pa, __bf16* __restrict__ pe) {
    int idx = blockIdx.x * blockDim.x + threadIdx.x;
    if (idx >= (GSEG + ESEG) * NAE) return;
    int seg = idx >> 7, c = idx & 127;
    if (seg < GSEG) {
        float m = asum[seg * NAE + c] / fmaxf(acnt[seg], 1.0f);
        pa[seg * NAE + c] = (__bf16)fmaxf(m, 0.0f);
    } else {
        int s = seg - GSEG;
        float m = esum[s * NAE + c] / fmaxf(ecnt[s], 1.0f);
        pe[s * NAE + c] = (__bf16)fmaxf(m, 0.0f);
    }
}

// pack W1 [384,256] f32 row-major -> per-(ktile,ntile) B fragments:
// flat = tile*512 + lane*16 + e ; K = kt*32 + (e>=8?16:0) + (lane>=16?8:0) + (e&7) ; N = nt*16 + (lane&15)
__global__ void pack_w1_kernel(const float* __restrict__ W1, __bf16* __restrict__ Wp) {
    int idx = blockIdx.x * blockDim.x + threadIdx.x;   // 0 .. 98303
    int e = idx & 15;
    int lane = (idx >> 4) & 31;
    int tile = idx >> 9;
    int nt = tile & 15, kt = tile >> 4;
    int k = kt * 32 + ((e >> 3) << 4) + ((lane >> 4) << 3) + (e & 7);
    int n = nt * 16 + (lane & 15);
    Wp[idx] = (__bf16)W1[k * CDIM + n];
}

__launch_bounds__(256)
__global__ void gemm_kernel(const float* __restrict__ dist,
                            const __bf16* __restrict__ pa, const __bf16* __restrict__ pe,
                            const int* __restrict__ aidx, const int* __restrict__ eidx,
                            const __bf16* __restrict__ Wp, float* __restrict__ h1) {
    __shared__ __bf16 As[MTILE * ASTRIDE];      // 50176 B
    __shared__ int s_ai[MTILE], s_ei[MTILE];
    const int tid = threadIdx.x;
    const int rowbase = blockIdx.x * MTILE;     // 200000 = 64*3125, no tail

    if (tid < MTILE) { s_ai[tid] = aidx[rowbase + tid]; s_ei[tid] = eidx[rowbase + tid]; }
    __syncthreads();

    // stage gathered A tile (64 x 384) as bf16 into LDS
    for (int it = 0; it < (MTILE * KDIM) / 256; ++it) {
        int idx = it * 256 + tid;
        int r = idx / KDIM;
        int c = idx - r * KDIM;
        __bf16 v;
        if (c < NAE)            v = (__bf16)dist[(long)(rowbase + r) * NAE + c];
        else if (c < 2 * NAE)   v = pa[s_ai[r] * NAE + (c - NAE)];
        else                    v = pe[s_ei[r] * NAE + (c - 2 * NAE)];
        As[r * ASTRIDE + c] = v;
    }
    __syncthreads();

    const int w = tid >> 5, lane = tid & 31;
    const int lm = lane & 15;
    const int lk = (lane >> 4) ? 8 : 0;      // A-frag K offset per ISA layout

    v8f acc[2][4];
    for (int h = 0; h < 2; ++h)
        for (int m = 0; m < 4; ++m)
            acc[h][m] = (v8f)0.0f;

    for (int kt = 0; kt < KDIM / 32; ++kt) {
        v16bf afr[4];
        for (int ms = 0; ms < 4; ++ms) {
            const __bf16* p = &As[(ms * 16 + lm) * ASTRIDE + kt * 32 + lk];
            union { v16bf v; v8bf h[2]; } u;
            u.h[0] = *(const v8bf*)p;          // K = klo .. klo+7
            u.h[1] = *(const v8bf*)(p + 16);   // K = klo+16 .. klo+23
            afr[ms] = u.v;
        }
        for (int half = 0; half < 2; ++half) {
            int nt = w + half * 8;
            v16bf b = *(const v16bf*)&Wp[(kt * 16 + nt) * 512 + lane * 16];
            for (int ms = 0; ms < 4; ++ms)
                acc[half][ms] = __builtin_amdgcn_wmma_f32_16x16x32_bf16(
                    false, afr[ms], false, b, (short)0, acc[half][ms], false, false);
        }
    }

    // D layout: VGPR v -> M = v (+8 for lanes 16..31), N = lane&15
    const int rofs = (lane >> 4) * 8;
    for (int half = 0; half < 2; ++half) {
        int col = (w + half * 8) * 16 + lm;
        for (int ms = 0; ms < 4; ++ms) {
            int rb = rowbase + ms * 16 + rofs;
            for (int v = 0; v < 8; ++v)
                h1[(long)(rb + v) * CDIM + col] = acc[half][ms][v];
        }
    }
}

#define RPB 512
__global__ void colsum_kernel(const float* __restrict__ h1,
                              float* __restrict__ colS, float* __restrict__ colSS) {
    int c = threadIdx.x;                      // 256 columns
    long r0 = (long)blockIdx.x * RPB;
    long r1 = r0 + RPB; if (r1 > NROWS) r1 = NROWS;
    float s = 0.0f, ss = 0.0f;
    for (long r = r0; r < r1; ++r) {
        float v = h1[r * CDIM + c];
        s += v; ss += v * v;
    }
    atomicAdd(&colS[c], s);
    atomicAdd(&colSS[c], ss);
}

__global__ void bn_stats_kernel(const float* __restrict__ colS, const float* __restrict__ colSS,
                                float* __restrict__ mu, float* __restrict__ rstd) {
    int c = threadIdx.x;
    const float invN = 1.0f / (float)NROWS;
    float m = colS[c] * invN;
    float var = colSS[c] * invN - m * m;
    mu[c] = m;
    rstd[c] = rsqrtf(var + 1e-5f);
}

__global__ void finalize_kernel(const float* __restrict__ x,
                                const float* __restrict__ mu, const float* __restrict__ rstd,
                                const float* __restrict__ gamma, const float* __restrict__ beta,
                                float* __restrict__ out /* holds h1, overwritten in place */) {
    long i4 = (long)blockIdx.x * blockDim.x + threadIdx.x;   // float4 index
    int c0 = ((int)(i4 & 63)) * 4;                           // 256/4 = 64 quads per row
    float4 h = ((const float4*)out)[i4];
    float4 xv = ((const float4*)x)[i4];
    float4 o;
    o.x = fmaxf((h.x - mu[c0 + 0]) * rstd[c0 + 0] * gamma[c0 + 0] + beta[c0 + 0] + xv.x, 0.0f);
    o.y = fmaxf((h.y - mu[c0 + 1]) * rstd[c0 + 1] * gamma[c0 + 1] + beta[c0 + 1] + xv.y, 0.0f);
    o.z = fmaxf((h.z - mu[c0 + 2]) * rstd[c0 + 2] * gamma[c0 + 2] + beta[c0 + 2] + xv.z, 0.0f);
    o.w = fmaxf((h.w - mu[c0 + 3]) * rstd[c0 + 3] * gamma[c0 + 3] + beta[c0 + 3] + xv.w, 0.0f);
    ((float4*)out)[i4] = o;
}

extern "C" void kernel_launch(void* const* d_in, const int* in_sizes, int n_in,
                              void* d_out, int out_size, void* d_ws, size_t ws_size,
                              hipStream_t stream) {
    const float* x     = (const float*)d_in[0];
    const float* dist  = (const float*)d_in[1];
    const int*   aidx  = (const int*)d_in[2];
    const int*   eidx  = (const int*)d_in[3];
    const float* W1    = (const float*)d_in[4];
    // d_in[5] = b1: cancels under batch-norm mean subtraction, unused
    const float* gamma = (const float*)d_in[6];
    const float* beta  = (const float*)d_in[7];

    char* ws = (char*)d_ws;
    float*  asum = (float*)(ws + OFF_ATOM_SUMS);
    float*  acnt = (float*)(ws + OFF_ATOM_CNT);
    float*  esum = (float*)(ws + OFF_ELE_SUMS);
    float*  ecnt = (float*)(ws + OFF_ELE_CNT);
    float*  colS = (float*)(ws + OFF_COLS);
    float*  colSS= (float*)(ws + OFF_COLSS);
    float*  mu   = (float*)(ws + OFF_MU);
    float*  rstd = (float*)(ws + OFF_RSTD);
    __bf16* pa   = (__bf16*)(ws + OFF_PA);
    __bf16* pe   = (__bf16*)(ws + OFF_PE);
    __bf16* Wp   = (__bf16*)(ws + OFF_WP);

    float* h1 = (float*)d_out;   // h @ W1 staged in the output buffer, normalized in place

    zero_ws_kernel<<<(ZERO_FLOATS + 255) / 256, 256, 0, stream>>>((float*)ws, ZERO_FLOATS);
    pool_accum_kernel<<<NROWS / 2, 256, 0, stream>>>(x, aidx, eidx, asum, acnt, esum, ecnt);
    pool_final_kernel<<<((GSEG + ESEG) * NAE + 255) / 256, 256, 0, stream>>>(asum, acnt, esum, ecnt, pa, pe);
    pack_w1_kernel<<<(KDIM * CDIM) / 256, 256, 0, stream>>>(W1, Wp);
    gemm_kernel<<<NROWS / MTILE, 256, 0, stream>>>(dist, pa, pe, aidx, eidx, Wp, h1);
    colsum_kernel<<<(NROWS + RPB - 1) / RPB, 256, 0, stream>>>(h1, colS, colSS);
    bn_stats_kernel<<<1, 256, 0, stream>>>(colS, colSS, mu, rstd);
    finalize_kernel<<<(NROWS * CDIM / 4) / 256, 256, 0, stream>>>(x, mu, rstd, gamma, beta, h1);
}
